// SelfAttention_7859790152248
// MI455X (gfx1250) — matrix operand order repaired and checked by reading, not compiled, not run on hardware
//
#include <hip/hip_runtime.h>

typedef __attribute__((ext_vector_type(16))) __bf16 v16bf;
typedef __attribute__((ext_vector_type(8)))  float  v8f;
typedef __attribute__((ext_vector_type(4)))  unsigned int u32x4;

#define S_LEN 2048
#define D_DIM 768
#define B_NUM 8

// Workgroup GEMM tile: 128(M) x 64(N), K-step 32, 8 waves (4m x 2n),
// each wave owns a 32x32 accumulator = 2x2 WMMA 16x16 tiles.
#define BM 128
#define BN 64
#define BK 32
#define LDS_ELEMS ((BM + BN) * BK)   // bf16 elements per stage buffer (12 KB)

union FragBF {
  v16bf v;
  u32x4 u[2];
};

__device__ __forceinline__ __bf16 f2bf(float f) {
  union { float f; unsigned int u; } c; c.f = f;
  unsigned int r = c.u + 0x7FFFu + ((c.u >> 16) & 1u);   // round-to-nearest-even
  union { unsigned short s; __bf16 b; } o; o.s = (unsigned short)(r >> 16);
  return o.b;
}

__global__ void cvt_bf16_kernel(const float* __restrict__ src, __bf16* __restrict__ dst, int n) {
  int i = blockIdx.x * blockDim.x + threadIdx.x;
  int stride = gridDim.x * blockDim.x;
  for (; i < n; i += stride) dst[i] = f2bf(src[i]);
}

// ---- CDNA5 async global -> LDS copy (ASYNCcnt-tracked) --------------------
__device__ __forceinline__ void async_ld16(const __bf16* gptr, unsigned lds_off) {
  asm volatile("global_load_async_to_lds_b128 %0, %1, off"
               :: "v"(lds_off), "v"(gptr) : "memory");
}
__device__ __forceinline__ void wait_async0() {
  asm volatile("s_wait_asynccnt 0" ::: "memory");
}

// Stage A[BM x BK] (rows m0..) and B[BN x BK] (rows n0..) into LDS buffer.
// 256 threads: A = 512 x 16B chunks (2/thread), B = 256 x 16B chunks (1/thread).
__device__ __forceinline__ void stage_tiles(const __bf16* __restrict__ A, int lda,
                                            const __bf16* __restrict__ Bsrc, int ldb,
                                            int m0, int n0, int kk,
                                            unsigned ldsA, unsigned ldsB, int tid) {
  int c0 = tid, c1 = tid + 256;
  async_ld16(A + (size_t)(m0 + (c0 >> 2)) * lda + kk + (c0 & 3) * 8, ldsA + c0 * 16);
  async_ld16(A + (size_t)(m0 + (c1 >> 2)) * lda + kk + (c1 & 3) * 8, ldsA + c1 * 16);
  async_ld16(Bsrc + (size_t)(n0 + (tid >> 2)) * ldb + kk + (tid & 3) * 8, ldsB + tid * 16);
}

// Core: acc[i][j] (32x32 per wave) += A[m0+.., k] * Bsrc[n0+.., k]^T over K.
__device__ __forceinline__ void gemm_core(const __bf16* __restrict__ A, int lda,
                                          const __bf16* __restrict__ Bsrc, int ldb,
                                          int K, int m0, int n0,
                                          __bf16* smem, v8f acc[2][2]) {
  const int tid  = threadIdx.x;
  const int lane = tid & 31, wave = tid >> 5;
  const int l = lane & 15, h = lane >> 4;
  const int wm = wave >> 1, wn = wave & 1;

  const unsigned smbase = (unsigned)(unsigned long long)smem;  // flat low 32b = LDS byte offset
  const unsigned ldsA[2] = { smbase,               smbase + LDS_ELEMS * 2 };
  const unsigned ldsB[2] = { smbase + BM * BK * 2, smbase + LDS_ELEMS * 2 + BM * BK * 2 };

  for (int i = 0; i < 2; ++i)
    for (int j = 0; j < 2; ++j)
      acc[i][j] = {};

  const int nk = K / BK;
  stage_tiles(A, lda, Bsrc, ldb, m0, n0, 0, ldsA[0], ldsB[0], tid);
  wait_async0();
  __syncthreads();

  for (int kb = 0; kb < nk; ++kb) {
    const int cur = kb & 1;
    if (kb + 1 < nk)
      stage_tiles(A, lda, Bsrc, ldb, m0, n0, (kb + 1) * BK, ldsA[cur ^ 1], ldsB[cur ^ 1], tid);

    const __bf16* abuf = smem + cur * LDS_ELEMS;
    const __bf16* bbuf = abuf + BM * BK;
    FragBF a[2], b[2];
    for (int i = 0; i < 2; ++i) {
      const __bf16* ar = abuf + (size_t)(wm * 32 + i * 16 + l) * BK;
      a[i].u[0] = *reinterpret_cast<const u32x4*>(ar + 8 * h);
      a[i].u[1] = *reinterpret_cast<const u32x4*>(ar + 8 * h + 16);
      const __bf16* br = bbuf + (size_t)(wn * 32 + i * 16 + l) * BK;
      b[i].u[0] = *reinterpret_cast<const u32x4*>(br + 8 * h);
      b[i].u[1] = *reinterpret_cast<const u32x4*>(br + 8 * h + 16);
    }
    for (int i = 0; i < 2; ++i)
      for (int j = 0; j < 2; ++j)
        acc[i][j] = __builtin_amdgcn_wmma_f32_16x16x32_bf16(false, a[i].v, false, b[j].v,
                                                            (short)0, acc[i][j], false, false);
    wait_async0();     // next stage landed (no-op on last iter)
    __syncthreads();   // all waves done reading `cur` before it is restaged
  }
}

// Y[m,n] = sum_d Zb[m,d]*Wb[n,d] + bias[n]; optional bf16 Y ([M,D]) and/or
// per-batch transposed Yt ([b][D][S] bf16).
__global__ void proj_kernel(const __bf16* __restrict__ Zb, const __bf16* __restrict__ Wb,
                            const float* __restrict__ bias,
                            __bf16* __restrict__ Y, __bf16* __restrict__ Yt) {
  __shared__ __bf16 smem[2 * LDS_ELEMS];
  const int m0 = blockIdx.y * BM, n0 = blockIdx.x * BN;
  v8f acc[2][2];
  gemm_core(Zb, D_DIM, Wb, D_DIM, D_DIM, m0, n0, smem, acc);

  const int lane = threadIdx.x & 31, wave = threadIdx.x >> 5;
  const int l = lane & 15, h = lane >> 4;
  const int wm = wave >> 1, wn = wave & 1;
  for (int i = 0; i < 2; ++i)
    for (int j = 0; j < 2; ++j) {
      const int n = n0 + wn * 32 + j * 16 + l;
      const int mbase = m0 + wm * 32 + i * 16 + 8 * h;
      const float bv = bias[n];
      union { __bf16 t[8]; u32x4 uv; } tv;
      for (int r = 0; r < 8; ++r) {
        __bf16 o = f2bf(acc[i][j][r] + bv);
        tv.t[r] = o;
        if (Y) Y[(size_t)(mbase + r) * D_DIM + n] = o;
      }
      if (Yt) {
        const int b  = mbase / S_LEN;
        const int s0 = mbase % S_LEN;          // 8 consecutive sequence positions
        __bf16* dst = Yt + (size_t)b * D_DIM * S_LEN + (size_t)n * S_LEN + s0;
        *reinterpret_cast<u32x4*>(dst) = tv.uv;
      }
    }
}

// scores[q,k] = scale * sum_d Qb[q,d]*Kb[k,d]   (single batch, S x S fp32)
__global__ void scores_kernel(const __bf16* __restrict__ Qb, const __bf16* __restrict__ Kb,
                              float* __restrict__ scores, float scale) {
  __shared__ __bf16 smem[2 * LDS_ELEMS];
  const int m0 = blockIdx.y * BM, n0 = blockIdx.x * BN;
  v8f acc[2][2];
  gemm_core(Qb, D_DIM, Kb, D_DIM, D_DIM, m0, n0, smem, acc);

  const int lane = threadIdx.x & 31, wave = threadIdx.x >> 5;
  const int l = lane & 15, h = lane >> 4;
  const int wm = wave >> 1, wn = wave & 1;
  for (int i = 0; i < 2; ++i)
    for (int j = 0; j < 2; ++j) {
      const int n = n0 + wn * 32 + j * 16 + l;
      const int mbase = m0 + wm * 32 + i * 16 + 8 * h;
      for (int r = 0; r < 8; ++r)
        scores[(size_t)(mbase + r) * S_LEN + n] = acc[i][j][r] * scale;
    }
}

// Row softmax over S=2048, fp32 in -> bf16 out. One block per row.
__global__ void softmax_kernel(const float* __restrict__ scores, __bf16* __restrict__ attnb) {
  int q = blockIdx.x;
  const float* row = scores + (size_t)q * S_LEN;
  __shared__ float sm[256];
  int t = threadIdx.x;
  float mx = -3.0e38f;
  for (int i = t; i < S_LEN; i += 256) mx = fmaxf(mx, row[i]);
  sm[t] = mx; __syncthreads();
  for (int s = 128; s > 0; s >>= 1) { if (t < s) sm[t] = fmaxf(sm[t], sm[t + s]); __syncthreads(); }
  mx = sm[0]; __syncthreads();
  float sum = 0.f;
  for (int i = t; i < S_LEN; i += 256) sum += __expf(row[i] - mx);
  sm[t] = sum; __syncthreads();
  for (int s = 128; s > 0; s >>= 1) { if (t < s) sm[t] += sm[t + s]; __syncthreads(); }
  float inv = 1.0f / sm[0];
  __bf16* orow = attnb + (size_t)q * S_LEN;
  for (int i = t; i < S_LEN; i += 256) orow[i] = f2bf(__expf(row[i] - mx) * inv);
}

// out[q,e] = sum_k attn[q,k] * Vt[e,k]   (single batch; Vt is [D][S] bf16)
__global__ void out_kernel(const __bf16* __restrict__ attnb, const __bf16* __restrict__ Vtb,
                           float* __restrict__ out) {
  __shared__ __bf16 smem[2 * LDS_ELEMS];
  const int m0 = blockIdx.y * BM, n0 = blockIdx.x * BN;
  v8f acc[2][2];
  gemm_core(attnb, S_LEN, Vtb, S_LEN, S_LEN, m0, n0, smem, acc);

  const int lane = threadIdx.x & 31, wave = threadIdx.x >> 5;
  const int l = lane & 15, h = lane >> 4;
  const int wm = wave >> 1, wn = wave & 1;
  for (int i = 0; i < 2; ++i)
    for (int j = 0; j < 2; ++j) {
      const int n = n0 + wn * 32 + j * 16 + l;
      const int mbase = m0 + wm * 32 + i * 16 + 8 * h;
      for (int r = 0; r < 8; ++r)
        out[(size_t)(mbase + r) * D_DIM + n] = acc[i][j][r];
    }
}

extern "C" void kernel_launch(void* const* d_in, const int* in_sizes, int n_in,
                              void* d_out, int out_size, void* d_ws, size_t ws_size,
                              hipStream_t stream) {
  const float* Z  = (const float*)d_in[0];
  const float* Wq = (const float*)d_in[1];
  const float* bq = (const float*)d_in[2];
  const float* Wk = (const float*)d_in[3];
  const float* bk = (const float*)d_in[4];
  const float* Wv = (const float*)d_in[5];
  const float* bv = (const float*)d_in[6];
  float* out = (float*)d_out;

  char* ws = (char*)d_ws;
  size_t off = 0;
  auto alloc = [&](size_t bytes) -> char* {
    char* p = ws + off;
    off += (bytes + 255) & ~(size_t)255;
    return p;
  };
  const size_t BSD = (size_t)B_NUM * S_LEN * D_DIM;
  const size_t DD  = (size_t)D_DIM * D_DIM;

  __bf16* Zb  = (__bf16*)alloc(BSD * 2);
  __bf16* Wqb = (__bf16*)alloc(DD * 2);
  __bf16* Wkb = (__bf16*)alloc(DD * 2);
  __bf16* Wvb = (__bf16*)alloc(DD * 2);
  __bf16* Qb  = (__bf16*)alloc(BSD * 2);
  __bf16* Kb  = (__bf16*)alloc(BSD * 2);
  __bf16* Vtb = (__bf16*)alloc(BSD * 2);                    // [b][D][S]
  float*  scb = (float*)alloc((size_t)S_LEN * S_LEN * 4);   // reused per batch
  __bf16* atb = (__bf16*)alloc((size_t)S_LEN * S_LEN * 2);  // reused per batch
  (void)ws_size; (void)in_sizes; (void)n_in; (void)out_size;

  cvt_bf16_kernel<<<1024, 256, 0, stream>>>(Z,  Zb,  (int)BSD);
  cvt_bf16_kernel<<<256,  256, 0, stream>>>(Wq, Wqb, (int)DD);
  cvt_bf16_kernel<<<256,  256, 0, stream>>>(Wk, Wkb, (int)DD);
  cvt_bf16_kernel<<<256,  256, 0, stream>>>(Wv, Wvb, (int)DD);

  // Projections: M = B*S = 16384, N = D = 768.
  dim3 pgrid(D_DIM / BN, (unsigned)(BSD / D_DIM / BM));   // (12, 128)
  proj_kernel<<<pgrid, 256, 0, stream>>>(Zb, Wqb, bq, Qb, nullptr);
  proj_kernel<<<pgrid, 256, 0, stream>>>(Zb, Wkb, bk, Kb, nullptr);
  proj_kernel<<<pgrid, 256, 0, stream>>>(Zb, Wvb, bv, nullptr, Vtb);

  const float scale = 0.03608439182435161f;   // 1/sqrt(768)
  for (int b = 0; b < B_NUM; ++b) {
    const __bf16* Qbb = Qb + (size_t)b * S_LEN * D_DIM;
    const __bf16* Kbb = Kb + (size_t)b * S_LEN * D_DIM;
    const __bf16* Vtb_b = Vtb + (size_t)b * D_DIM * S_LEN;
    float* outb = out + (size_t)b * S_LEN * D_DIM;
    scores_kernel<<<dim3(S_LEN / BN, S_LEN / BM), 256, 0, stream>>>(Qbb, Kbb, scb, scale);
    softmax_kernel<<<S_LEN, 256, 0, stream>>>(scb, atb);
    out_kernel<<<dim3(D_DIM / BN, S_LEN / BM), 256, 0, stream>>>(atb, Vtb_b, outb);
  }
}
